// RelationAttention_68204080660552
// MI455X (gfx1250) — matrix operand order repaired and checked by reading, not compiled, not run on hardware
//
#include <hip/hip_runtime.h>
#include <hip/hip_bf16.h>

// ---------------------------------------------------------------------------
// Fused graph attention for MI455X (gfx1250, wave32, WMMA).
// GEMMs run as v_wmma_f32_16x16x32_f16 (f16 inputs, f32 accumulate) because
// f32 WMMA (16x16x4) would leave the kernel ~4x compute-bound vs the
// 23.3 TB/s HBM roofline; everything nonlinear stays f32.
// ---------------------------------------------------------------------------

typedef __attribute__((ext_vector_type(16))) _Float16 v16h;
typedef __attribute__((ext_vector_type(8)))  _Float16 v8h;
typedef __attribute__((ext_vector_type(8)))  float    v8f;

#define D       128
#define H       4
#define DKH     32
#define TILE_M  16
#define WAVES_PER_WG 16
#define WG_THREADS   512
#define SCR_STRIDE   132   // padded row stride (floats) -> no LDS bank conflicts

// ---- monotone float<->uint mapping so atomicMax(uint) == float max --------
__device__ __forceinline__ unsigned f2key(float f) {
  unsigned b = __float_as_uint(f);
  return (b & 0x80000000u) ? ~b : (b | 0x80000000u);
}
__device__ __forceinline__ float key2f(unsigned k) {
  unsigned b = (k & 0x80000000u) ? (k ^ 0x80000000u) : ~k;
  return __uint_as_float(b);
}

// ---- A fragment: 16-bit A 16x32 layout (ISA 7.12.2).
// lane group g = lane>>4:  halves[0..7] = A[m][8g+j],  halves[8..15] = A[m][16+8g+j]
__device__ __forceinline__ v16h load_a_frag(const float* __restrict__ p) {
  v16h a;
#pragma unroll
  for (int j = 0; j < 8; ++j) a[j]     = (_Float16)p[j];
#pragma unroll
  for (int j = 0; j < 8; ++j) a[8 + j] = (_Float16)p[16 + j];
  return a;
}

// ---- B fragment from pre-swizzled LDS weights: 16 contiguous halves -------
__device__ __forceinline__ v16h load_b_frag(const _Float16* __restrict__ swz,
                                            int kb, int t, int lane) {
  const _Float16* p = swz + (((kb * 8 + t) * 32 + lane) << 4);
  v8h lo = *(const v8h*)(p);
  v8h hi = *(const v8h*)(p + 8);
  v16h b;
#pragma unroll
  for (int j = 0; j < 8; ++j) { b[j] = lo[j]; b[8 + j] = hi[j]; }
  return b;
}

// ---- convert a [Kdim x 128] f32 weight into f16 WMMA-B fragment order -----
// frag(kb,t,lane)[j] = W[kb*32 + 16*(j>>3) + 8*(lane>>4) + (j&7)][t*16 + (lane&15)]
__device__ __forceinline__ void stage_weight(const float* __restrict__ W,
                                             _Float16* __restrict__ swz,
                                             int Kdim, int tid, int nthr) {
  int total = Kdim * D;
  for (int idx = tid; idx < total; idx += nthr) {
    int j  = idx & 15;
    int L  = (idx >> 4) & 31;
    int t  = (idx >> 9) & 7;
    int kb = idx >> 12;
    int srcRow = kb * 32 + ((j >> 3) << 4) + ((L >> 4) << 3) + (j & 7);
    int srcCol = t * 16 + (L & 15);
    swz[idx] = (_Float16)W[srcRow * D + srcCol];
  }
}

// ===========================================================================
// Kernel 1: zero aggregated output + segment-max keys + denominators
// ===========================================================================
__global__ void k_init(float* __restrict__ agg, unsigned* __restrict__ smaxKeys,
                       float* __restrict__ denWS, int ND, int NH) {
  int i = blockIdx.x * blockDim.x + threadIdx.x;
  if (i < ND) agg[i] = 0.0f;
  if (i < NH) { smaxKeys[i] = 0u; denWS[i] = 0.0f; }
}

// ===========================================================================
// Kernel 2: per-edge K GEMM -> scores (+segment max), MLP hidden GEMM ->
//           relu -> W2 GEMV -> weight_preds.  One wave per 16-edge tile.
// ===========================================================================
__global__ __launch_bounds__(WG_THREADS, 1)
void k_edge_gemm(const float* __restrict__ h_src, const float* __restrict__ Q,
                 const int* __restrict__ dst_idx,
                 const float* __restrict__ Wk, const float* __restrict__ W1,
                 const float* __restrict__ b1, const float* __restrict__ W2,
                 const float* __restrict__ b2,
                 float* __restrict__ scoresWS, unsigned* __restrict__ smaxKeys,
                 float* __restrict__ wp_out, int numTiles) {
  extern __shared__ char smem[];
  _Float16* wk_swz = (_Float16*)(smem);            // 32 KB
  _Float16* w1_swz = (_Float16*)(smem + 32768);    // 64 KB
  float*    b1s    = (float*)(smem + 98304);       // 512 B
  float*    w2s    = (float*)(smem + 98816);       // 512 B
  float*    scratch= (float*)(smem + 99328);       // 16 waves * 16*132 f

  const int tid = threadIdx.x;
  stage_weight(Wk, wk_swz, D, tid, WG_THREADS);
  stage_weight(W1, w1_swz, 2 * D, tid, WG_THREADS);
  if (tid < D) { b1s[tid] = b1[tid]; w2s[tid] = W2[tid]; }
  __syncthreads();

  const int w = tid >> 5, lane = tid & 31;
  const int g = lane >> 4, c = lane & 15;
  const int tile = blockIdx.x * WAVES_PER_WG + w;
  if (tile >= numTiles) return;

  float* wscr = scratch + w * (TILE_M * SCR_STRIDE);
  const int  arowIdx = tile * TILE_M + c;                  // A-row fed by this lane
  const float* arow  = h_src + (size_t)arowIdx * D;
  const float* qrow  = Q     + (size_t)arowIdx * D;

  v8f zero = {};
  v8f acc[8];

  // ---------------- K = h_src @ Wk ------------------------------------
#pragma unroll
  for (int t = 0; t < 8; ++t) acc[t] = zero;
#pragma unroll
  for (int kb = 0; kb < 4; ++kb) {
    v16h a = load_a_frag(arow + kb * 32 + 8 * g);
#pragma unroll
    for (int t = 0; t < 8; ++t) {
      v16h b = load_b_frag(wk_swz, kb, t, lane);
      acc[t] = __builtin_amdgcn_wmma_f32_16x16x32_f16(false, a, false, b,
                                                      (short)0, acc[t], false, false);
    }
  }
  // dump K tile: C/D layout -> row = 8g+r, col = 16t+c
#pragma unroll
  for (int t = 0; t < 8; ++t)
#pragma unroll
    for (int r = 0; r < 8; ++r)
      wscr[(g * 8 + r) * SCR_STRIDE + t * 16 + c] = acc[t][r];
  asm volatile("s_wait_dscnt 0" ::: "memory");

  // ---------------- scores + segment max ------------------------------
#pragma unroll
  for (int pp = 0; pp < 2; ++pp) {
    int p  = lane + 32 * pp;                 // 64 (edge,head) pairs / wave
    int el = p >> 2, h = p & 3;
    int edge = tile * TILE_M + el;
    const float* q  = Q + (size_t)edge * D + h * DKH;
    const float* kk = wscr + el * SCR_STRIDE + h * DKH;
    float s = 0.0f;
#pragma unroll
    for (int i = 0; i < DKH; ++i) s = fmaf(q[i], kk[i], s);
    s *= 0.17677669529663688f;               // 1/sqrt(32)
    scoresWS[edge * H + h] = s;
    atomicMax(&smaxKeys[(size_t)dst_idx[edge] * H + h], f2key(s));
  }
  asm volatile("s_wait_dscnt 0" ::: "memory");

  // ---------------- hid = relu([h_src | Q] @ W1 + b1) ------------------
#pragma unroll
  for (int t = 0; t < 8; ++t) acc[t] = zero;
#pragma unroll
  for (int kb = 0; kb < 8; ++kb) {
    const float* src = (kb < 4) ? (arow + kb * 32) : (qrow + (kb - 4) * 32);
    v16h a = load_a_frag(src + 8 * g);
#pragma unroll
    for (int t = 0; t < 8; ++t) {
      v16h b = load_b_frag(w1_swz, kb, t, lane);
      acc[t] = __builtin_amdgcn_wmma_f32_16x16x32_f16(false, a, false, b,
                                                      (short)0, acc[t], false, false);
    }
  }
#pragma unroll
  for (int t = 0; t < 8; ++t)
#pragma unroll
    for (int r = 0; r < 8; ++r) {
      int cc = t * 16 + c;
      float v = acc[t][r] + b1s[cc];
      wscr[(g * 8 + r) * SCR_STRIDE + cc] = v > 0.0f ? v : 0.0f;
    }
  asm volatile("s_wait_dscnt 0" ::: "memory");

  // ---------------- weight_preds = hid @ W2 + b2 -----------------------
  if (lane < 16) {
    float s = b2[0];
    const float* hrow = wscr + lane * SCR_STRIDE;
#pragma unroll 8
    for (int i = 0; i < D; ++i) s = fmaf(hrow[i], w2s[i], s);
    wp_out[tile * TILE_M + lane] = s;
  }
}

// ===========================================================================
// Kernel 3: ex = exp(score - segmax[dst]); den[dst] += ex
// ===========================================================================
__global__ void k_softmax_edges(const float* __restrict__ scoresWS,
                                const unsigned* __restrict__ smaxKeys,
                                const int* __restrict__ dst_idx,
                                float* __restrict__ exWS,
                                float* __restrict__ denWS, int EH) {
  int i = blockIdx.x * blockDim.x + threadIdx.x;
  if (i >= EH) return;
  int e = i >> 2, h = i & 3;
  size_t seg = (size_t)dst_idx[e] * H + h;
  float m  = key2f(smaxKeys[seg]);
  float ex = __expf(scoresWS[i] - m);
  exWS[i] = ex;
  atomicAdd(&denWS[seg], ex);
}

// ===========================================================================
// Kernel 4: attn = ex/den[dst]; V = h_src @ Wv (WMMA, recomputed — same HBM
//           read as storing V but no 256 MB workspace); agg[dst] += V*attn
// ===========================================================================
__global__ __launch_bounds__(WG_THREADS, 1)
void k_aggregate(const float* __restrict__ h_src, const int* __restrict__ dst_idx,
                 const float* __restrict__ Wv,
                 const float* __restrict__ exWS, const float* __restrict__ denWS,
                 float* __restrict__ attn_out, float* __restrict__ agg_out,
                 int numTiles) {
  extern __shared__ char smem[];
  _Float16* wv_swz   = (_Float16*)(smem);        // 32 KB
  float*    attn_lds = (float*)(smem + 32768);   // 16 waves * 64 f

  const int tid = threadIdx.x;
  stage_weight(Wv, wv_swz, D, tid, WG_THREADS);
  __syncthreads();

  const int w = tid >> 5, lane = tid & 31;
  const int g = lane >> 4, c = lane & 15;
  const int tile = blockIdx.x * WAVES_PER_WG + w;
  if (tile >= numTiles) return;

  float* alds = attn_lds + w * 64;
#pragma unroll
  for (int pp = 0; pp < 2; ++pp) {
    int p  = lane + 32 * pp;
    int el = p >> 2, h = p & 3;
    int edge = tile * TILE_M + el;
    float a = exWS[edge * H + h] / denWS[(size_t)dst_idx[edge] * H + h];
    attn_out[edge * H + h] = a;
    alds[p] = a;
  }
  asm volatile("s_wait_dscnt 0" ::: "memory");

  const int  arowIdx = tile * TILE_M + c;
  const float* arow  = h_src + (size_t)arowIdx * D;
  v8f zero = {};
  v8f acc[8];
#pragma unroll
  for (int t = 0; t < 8; ++t) acc[t] = zero;
#pragma unroll
  for (int kb = 0; kb < 4; ++kb) {
    v16h a = load_a_frag(arow + kb * 32 + 8 * g);
#pragma unroll
    for (int t = 0; t < 8; ++t) {
      v16h b = load_b_frag(wv_swz, kb, t, lane);
      acc[t] = __builtin_amdgcn_wmma_f32_16x16x32_f16(false, a, false, b,
                                                      (short)0, acc[t], false, false);
    }
  }
#pragma unroll
  for (int r = 0; r < 8; ++r) {
    int mrow = g * 8 + r;
    int edge = tile * TILE_M + mrow;
    int dd   = dst_idx[edge];
    float* aggrow = agg_out + (size_t)dd * D;
#pragma unroll
    for (int t = 0; t < 8; ++t) {          // col 16t+c, head = t>>1
      float val = acc[t][r] * alds[mrow * H + (t >> 1)];
      atomicAdd(&aggrow[t * 16 + c], val);
    }
  }
}

// ===========================================================================
extern "C" void kernel_launch(void* const* d_in, const int* in_sizes, int n_in,
                              void* d_out, int out_size, void* d_ws, size_t ws_size,
                              hipStream_t stream) {
  const float* h_src = (const float*)d_in[0];
  const float* Q     = (const float*)d_in[1];
  const int*   dst_i = (const int*)d_in[3];
  const float* Wk    = (const float*)d_in[5];
  const float* Wv    = (const float*)d_in[6];
  const float* W1    = (const float*)d_in[7];
  const float* b1    = (const float*)d_in[8];
  const float* W2    = (const float*)d_in[9];
  const float* b2    = (const float*)d_in[10];

  const int E = in_sizes[2];                       // src_idx count == #edges
  const int N = (out_size - 5 * E) / D;            // out = N*D + E*H + E

  float* agg  = (float*)d_out;
  float* attn = agg + (size_t)N * D;
  float* wp   = attn + (size_t)E * H;

  float*    scoresWS = (float*)d_ws;
  float*    exWS     = scoresWS + (size_t)E * H;
  float*    denWS    = exWS + (size_t)E * H;
  unsigned* smaxKeys = (unsigned*)(denWS + (size_t)N * H);

  const int numTiles = E / TILE_M;                 // E = 500000 -> 31250 tiles
  const int ND = N * D, NH = N * H, EH = E * H;

  k_init<<<(ND + 255) / 256, 256, 0, stream>>>(agg, smaxKeys, denWS, ND, NH);

  const int gridT = (numTiles + WAVES_PER_WG - 1) / WAVES_PER_WG;
  const size_t lds2 = 32768 + 65536 + 512 + 512 +
                      (size_t)WAVES_PER_WG * TILE_M * SCR_STRIDE * 4;   // ~234 KB
  k_edge_gemm<<<gridT, WG_THREADS, lds2, stream>>>(h_src, Q, dst_i, Wk, W1, b1,
                                                   W2, b2, scoresWS, smaxKeys,
                                                   wp, numTiles);

  k_softmax_edges<<<(EH + 255) / 256, 256, 0, stream>>>(scoresWS, smaxKeys,
                                                        dst_i, exWS, denWS, EH);

  const size_t lds4 = 32768 + (size_t)WAVES_PER_WG * 64 * 4;
  k_aggregate<<<gridT, WG_THREADS, lds4, stream>>>(h_src, dst_i, Wv, exWS,
                                                   denWS, attn, agg, numTiles);
}
